// NonLocalBlock2D_81415400063436
// MI455X (gfx1250) — matrix operand order repaired and checked by reading, not compiled, not run on hardware
//
#include <hip/hip_runtime.h>

#define BB  8
#define CD  256
#define ICD 128
#define NT  4096

typedef __attribute__((ext_vector_type(16))) __bf16 v16bf;
typedef __attribute__((ext_vector_type(8)))  float  v8f;

union Frag { v16bf v; unsigned int u[8]; };

__device__ __forceinline__ v8f wmma_bf16(const Frag& a, const Frag& b, v8f c) {
  return __builtin_amdgcn_wmma_f32_16x16x32_bf16(false, a.v, false, b.v,
                                                 (short)0, c, false, false);
}

// A fragment (16x32 bf16, M x K), row-major source with leading dim ld.
// ISA 7.12.2: lane L holds row m=L&15; K pairs {h*8+0..7} then {h*8+16..23}, h=L>>4.
__device__ __forceinline__ void load_a(Frag& f, const __bf16* base, int ld, int lane) {
  const int m = lane & 15, h = lane >> 4;
  const __bf16* r = base + m * ld + h * 8;
#pragma unroll
  for (int v = 0; v < 4; ++v) f.u[v]     = *(const unsigned int*)(r + 2 * v);
#pragma unroll
  for (int v = 0; v < 4; ++v) f.u[4 + v] = *(const unsigned int*)(r + 16 + 2 * v);
}

// B fragment (32x16 bf16, K x N) loaded from B^T (N x K row-major, leading dim ld).
__device__ __forceinline__ void load_bt(Frag& f, const __bf16* baseT, int ld, int lane) {
  const int n = lane & 15, h = lane >> 4;
  const __bf16* r = baseT + n * ld + h * 8;
#pragma unroll
  for (int v = 0; v < 4; ++v) f.u[v]     = *(const unsigned int*)(r + 2 * v);
#pragma unroll
  for (int v = 0; v < 4; ++v) f.u[4 + v] = *(const unsigned int*)(r + 16 + 2 * v);
}

// ---- CDNA5 async global->LDS copy (ASYNCcnt), ISA ch.10 / 15.18.3 ----
// VDST VGPR holds the LDS byte offset (flat LDS aperture keeps offset in low 32 bits).
__device__ __forceinline__ unsigned lds_off(const void* p) {
  return (unsigned)(unsigned long long)p;
}
__device__ __forceinline__ void async_copy_b128(unsigned ldsoff, const void* gptr) {
  asm volatile("global_load_async_to_lds_b128 %0, %1, off"
               :: "v"(ldsoff), "v"(gptr) : "memory");
}
__device__ __forceinline__ void wait_async0() {
  asm volatile("s_wait_asynccnt 0" ::: "memory");
}

// ---------- Kernel 1a: x [B,C,N] f32 -> xT [B,N,C] bf16 (LDS-tiled transpose) ----
__global__ void __launch_bounds__(256)
k_convert_x(const float* __restrict__ x, __bf16* __restrict__ xT) {
  __shared__ __bf16 t[32][33];
  int bid = blockIdx.x;                      // BB * (CD/32) * (NT/32) = 8192
  int b = bid / (8 * 128);
  int r = bid - b * 8 * 128;
  int c0 = (r >> 7) * 32;
  int n0 = (r & 127) * 32;
  int lx = threadIdx.x & 31;
  int ly = threadIdx.x >> 5;                 // 8 rows per pass
#pragma unroll
  for (int i = 0; i < 32; i += 8)
    t[ly + i][lx] = (__bf16)x[(size_t)b * CD * NT + (size_t)(c0 + ly + i) * NT + n0 + lx];
  __syncthreads();
#pragma unroll
  for (int i = 0; i < 32; i += 8)
    xT[(size_t)b * NT * CD + (size_t)(n0 + ly + i) * CD + c0 + lx] = t[lx][ly + i];
}

// ---------- Kernel 1b: weights f32 -> bf16 ----------
__global__ void k_convert_w(const float* __restrict__ wg, const float* __restrict__ wt,
                            const float* __restrict__ wp, const float* __restrict__ wm,
                            __bf16* __restrict__ wgb, __bf16* __restrict__ wtb,
                            __bf16* __restrict__ wpb, __bf16* __restrict__ wmb) {
  int idx = blockIdx.x * blockDim.x + threadIdx.x;
  const int per = ICD * CD;
  if (idx >= 4 * per) return;
  int which = idx / per, o = idx - which * per;
  const float* s = which == 0 ? wg : which == 1 ? wt : which == 2 ? wp : wm;
  __bf16*      d = which == 0 ? wgb : which == 1 ? wtb : which == 2 ? wpb : wmb;
  d[o] = (__bf16)s[o];
}

// ---------- Kernel 2: projections g/theta/phi ----------
__global__ void __launch_bounds__(256)
k_proj(const __bf16* __restrict__ xT,
       const __bf16* __restrict__ wgb, const __bf16* __restrict__ wtb,
       const __bf16* __restrict__ wpb,
       const float* __restrict__ bg, const float* __restrict__ bt,
       const float* __restrict__ bp,
       __bf16* __restrict__ g, __bf16* __restrict__ theta, __bf16* __restrict__ phiT) {
  const int lane = threadIdx.x & 31;
  const int wid  = threadIdx.x >> 5;
  int t = blockIdx.x * 8 + wid;            // 3*8*8*64 = 12288 wave tiles
  int p = t / 4096; t -= p * 4096;
  int b = t / 512;  t -= b * 512;
  int m0 = (t >> 6) * 16;
  int n0 = (t & 63) * 64;

  const __bf16* W    = p == 0 ? wgb : p == 1 ? wtb : wpb;
  const float*  bias = p == 0 ? bg  : p == 1 ? bt  : bp;

  v8f zz = {};
  v8f acc[4] = {zz, zz, zz, zz};
  const __bf16* Bbase = xT + (size_t)b * NT * CD + (size_t)n0 * CD;
#pragma unroll
  for (int k0 = 0; k0 < CD; k0 += 32) {
    Frag a; load_a(a, W + m0 * CD + k0, CD, lane);
#pragma unroll
    for (int nt = 0; nt < 4; ++nt) {
      Frag bf; load_bt(bf, Bbase + nt * 16 * CD + k0, CD, lane);
      acc[nt] = wmma_bf16(a, bf, acc[nt]);
    }
  }
  const int h = lane >> 4, col = lane & 15;
  float bv[8];
#pragma unroll
  for (int v = 0; v < 8; ++v) bv[v] = bias[m0 + v + 8 * h];
#pragma unroll
  for (int nt = 0; nt < 4; ++nt) {
#pragma unroll
    for (int v = 0; v < 8; ++v) {
      float val = acc[nt][v] + bv[v];
      val = val > 0.f ? val : 0.f;
      int m = m0 + v + 8 * h;
      int n = n0 + nt * 16 + col;
      if (p == 0)
        g[(size_t)b * ICD * NT + (size_t)m * NT + n] = (__bf16)val;
      else if (p == 1)
        theta[(size_t)b * NT * ICD + (size_t)n * ICD + m] = (__bf16)val;
      else
        phiT[(size_t)b * NT * ICD + (size_t)n * ICD + m] = (__bf16)val;
    }
  }
}

// ---------- Kernel 3: flash attention, async double-buffered K/V in LDS ----------
// Block = 4 waves = 64 queries; all waves share the staged 32-key phi/g tiles.
__global__ void __launch_bounds__(128)
k_attn(const __bf16* __restrict__ theta, const __bf16* __restrict__ phiT,
       const __bf16* __restrict__ g, __bf16* __restrict__ ao) {
  __shared__ __align__(16) __bf16 kbuf[2][32 * ICD];  // [keys 32][IC 128]
  __shared__ __align__(16) __bf16 vbuf[2][ICD * 32];  // [IC 128][keys 32]
  __shared__ __align__(16) __bf16 pbuf[4][16 * 32];

  const int tid  = threadIdx.x;
  const int lane = tid & 31;
  const int wid  = tid >> 5;
  const int blk  = blockIdx.x;             // 512 blocks
  const int b    = blk >> 6;
  const int q0   = (blk & 63) * 64 + wid * 16;

  const __bf16* phb = phiT + (size_t)b * NT * ICD;   // [N][IC]
  const __bf16* gbp = g    + (size_t)b * ICD * NT;   // [IC][N]

  // Stage keys [j0, j0+32) into buffer bi: 8 async b128 per thread total.
  auto stage = [&](int j0, int bi) {
    const char* ksrc = (const char*)(phb + (size_t)j0 * ICD);   // 8KB contiguous
    unsigned kdst = lds_off(&kbuf[bi][0]);
#pragma unroll
    for (int s = 0; s < 4; ++s) {
      unsigned byteoff = (unsigned)(tid * 16 + s * 2048);
      async_copy_b128(kdst + byteoff, ksrc + byteoff);
    }
    const char* vsrc = (const char*)(gbp + (size_t)tid * NT + j0);  // row tid, 64B
    unsigned vdst = lds_off(&vbuf[bi][0]) + (unsigned)(tid * 64);
#pragma unroll
    for (int s = 0; s < 4; ++s)
      async_copy_b128(vdst + s * 16, vsrc + s * 16);
  };

  Frag qa[4];
  const __bf16* th = theta + (size_t)b * NT * ICD + (size_t)q0 * ICD;
#pragma unroll
  for (int c = 0; c < 4; ++c) load_a(qa[c], th + c * 32, ICD, lane);

  v8f zz = {};
  v8f o[8] = {zz, zz, zz, zz, zz, zz, zz, zz};
  float mrow[8], lrow[8];
#pragma unroll
  for (int v = 0; v < 8; ++v) { mrow[v] = -1e30f; lrow[v] = 0.f; }

  __bf16* myp = &pbuf[wid][0];
  const int h = lane >> 4, col = lane & 15;

  stage(0, 0);
  wait_async0();
  __syncthreads();

  for (int j0 = 0; j0 < NT; j0 += 32) {
    const int cur = (j0 >> 5) & 1;
    if (j0 + 32 < NT) stage(j0 + 32, cur ^ 1);   // overlap DMA with compute

    const __bf16* kb = &kbuf[cur][0];
    v8f s0 = zz, s1 = zz;
#pragma unroll
    for (int c = 0; c < 4; ++c) {
      Frag b0; load_bt(b0, kb + c * 32, ICD, lane);
      s0 = wmma_bf16(qa[c], b0, s0);
      Frag b1; load_bt(b1, kb + 16 * ICD + c * 32, ICD, lane);
      s1 = wmma_bf16(qa[c], b1, s1);
    }
    float p0[8], p1[8];
#pragma unroll
    for (int v = 0; v < 8; ++v) {
      float tm = fmaxf(s0[v], s1[v]);
#pragma unroll
      for (int msk = 1; msk < 16; msk <<= 1)
        tm = fmaxf(tm, __shfl_xor(tm, msk, 32));
      float nm = fmaxf(mrow[v], tm);
      float sc = __expf(mrow[v] - nm);
      mrow[v] = nm;
      p0[v] = __expf(s0[v] - nm);
      p1[v] = __expf(s1[v] - nm);
      float rs = p0[v] + p1[v];
#pragma unroll
      for (int msk = 1; msk < 16; msk <<= 1)
        rs += __shfl_xor(rs, msk, 32);
      lrow[v] = lrow[v] * sc + rs;
#pragma unroll
      for (int c = 0; c < 8; ++c) o[c][v] *= sc;
    }
    // C-fragment -> A-fragment re-layout via per-wave LDS (DS in order per wave)
#pragma unroll
    for (int v = 0; v < 8; ++v) {
      int row = v + 8 * h;
      myp[row * 32 + col]      = (__bf16)p0[v];
      myp[row * 32 + 16 + col] = (__bf16)p1[v];
    }
    Frag pa; load_a(pa, myp, 32, lane);
    const __bf16* vb = &vbuf[cur][0];
#pragma unroll
    for (int c = 0; c < 8; ++c) {
      Frag gf; load_bt(gf, vb + (c * 16) * 32, 32, lane);
      o[c] = wmma_bf16(pa, gf, o[c]);
    }
    wait_async0();      // my async issues for the next buffer are done
    __syncthreads();    // everyone's are; safe to read next, overwrite cur later
  }

  float rl[8];
#pragma unroll
  for (int v = 0; v < 8; ++v) rl[v] = 1.0f / lrow[v];
  __bf16* aob = ao + (size_t)b * NT * ICD + (size_t)q0 * ICD;
#pragma unroll
  for (int c = 0; c < 8; ++c) {
#pragma unroll
    for (int v = 0; v < 8; ++v) {
      int row = v + 8 * h;
      aob[(size_t)row * ICD + c * 16 + col] = (__bf16)(o[c][v] * rl[v]);
    }
  }
}

// ---------- Kernel 4: mask = relu(wm @ out + bm); d_out = mask + x ----------
__global__ void __launch_bounds__(256)
k_final(const __bf16* __restrict__ wmb, const float* __restrict__ bm,
        const __bf16* __restrict__ ao, const float* __restrict__ x,
        float* __restrict__ out) {
  const int lane = threadIdx.x & 31;
  const int wid  = threadIdx.x >> 5;
  int t = blockIdx.x * 8 + wid;            // 8192 wave tiles
  int b = t >> 10; t &= 1023;
  int c0 = (t >> 6) * 16;
  int n0 = (t & 63) * 64;

  v8f zz = {};
  v8f acc[4] = {zz, zz, zz, zz};
  const __bf16* Bbase = ao + (size_t)b * NT * ICD + (size_t)n0 * ICD;
#pragma unroll
  for (int k0 = 0; k0 < ICD; k0 += 32) {
    Frag a; load_a(a, wmb + c0 * ICD + k0, ICD, lane);
#pragma unroll
    for (int nt = 0; nt < 4; ++nt) {
      Frag bf; load_bt(bf, Bbase + nt * 16 * ICD + k0, ICD, lane);
      acc[nt] = wmma_bf16(a, bf, acc[nt]);
    }
  }
  const int h = lane >> 4, col = lane & 15;
  float bv[8];
#pragma unroll
  for (int v = 0; v < 8; ++v) bv[v] = bm[c0 + v + 8 * h];
  const float* xb = x   + (size_t)b * CD * NT;
  float*       ob = out + (size_t)b * CD * NT;
#pragma unroll
  for (int nt = 0; nt < 4; ++nt) {
#pragma unroll
    for (int v = 0; v < 8; ++v) {
      int cc = c0 + v + 8 * h;
      int n  = n0 + nt * 16 + col;
      float val = acc[nt][v] + bv[v];
      val = val > 0.f ? val : 0.f;
      ob[(size_t)cc * NT + n] = val + xb[(size_t)cc * NT + n];
    }
  }
}

extern "C" void kernel_launch(void* const* d_in, const int* in_sizes, int n_in,
                              void* d_out, int out_size, void* d_ws, size_t ws_size,
                              hipStream_t stream) {
  const float* x  = (const float*)d_in[0];
  const float* wg = (const float*)d_in[1];
  const float* bg = (const float*)d_in[2];
  const float* wt = (const float*)d_in[3];
  const float* bt = (const float*)d_in[4];
  const float* wp = (const float*)d_in[5];
  const float* bp = (const float*)d_in[6];
  const float* wm = (const float*)d_in[7];
  const float* bm = (const float*)d_in[8];

  char* ws = (char*)d_ws;
  size_t off = 0;
  __bf16* xT  = (__bf16*)(ws + off); off += (size_t)BB * NT * CD * 2;   // 16 MB
  __bf16* wgb = (__bf16*)(ws + off); off += (size_t)ICD * CD * 2;
  __bf16* wtb = (__bf16*)(ws + off); off += (size_t)ICD * CD * 2;
  __bf16* wpb = (__bf16*)(ws + off); off += (size_t)ICD * CD * 2;
  __bf16* wmb = (__bf16*)(ws + off); off += (size_t)CD * ICD * 2;
  __bf16* gB  = (__bf16*)(ws + off); off += (size_t)BB * ICD * NT * 2;  // 8 MB
  __bf16* thB = (__bf16*)(ws + off); off += (size_t)BB * NT * ICD * 2;  // 8 MB
  __bf16* phB = (__bf16*)(ws + off); off += (size_t)BB * NT * ICD * 2;  // 8 MB
  __bf16* aoB = xT;  // xT dead after k_proj; reuse for attention output

  k_convert_x<<<BB * (CD / 32) * (NT / 32), 256, 0, stream>>>(x, xT);
  k_convert_w<<<(4 * ICD * CD + 255) / 256, 256, 0, stream>>>(
      wg, wt, wp, wm, wgb, wtb, wpb, wmb);
  k_proj<<<1536, 256, 0, stream>>>(xT, wgb, wtb, wpb, bg, bt, bp, gB, thB, phB);
  k_attn<<<512, 128, 0, stream>>>(thB, phB, gB, aoB);
  k_final<<<1024, 256, 0, stream>>>(wmb, bm, aoB, x, (float*)d_out);
}
